// DynamicMoELayer_35553739276864
// MI455X (gfx1250) — matrix operand order repaired and checked by reading, not compiled, not run on hardware
//
#include <hip/hip_runtime.h>
#include <hip/hip_bf16.h>

#define NTOK 8192
#define HDIM 1024
#define FDIM 4096
#define NEXP 8
#define KSEL 1280   // NTOK * 1.25 / NEXP

typedef __attribute__((ext_vector_type(16))) __bf16 v16bf;
typedef __attribute__((ext_vector_type(8)))  __bf16 v8bf;
typedef __attribute__((ext_vector_type(8)))  float  v8f;

__device__ __forceinline__ unsigned f2bfu(float x) {
    // round-to-nearest-even fp32 -> bf16 (result in low 16 bits)
    unsigned u = __float_as_uint(x);
    return (u + 0x7FFFu + ((u >> 16) & 1u)) >> 16;
}

// ---------------------------------------------------------------------------
// Kernel 1: gate logits, written transposed [E, N]
// ---------------------------------------------------------------------------
__global__ __launch_bounds__(256) void gate_logits_kernel(
    const float* __restrict__ x, const float* __restrict__ gw,
    float* __restrict__ logitsT)
{
    __shared__ float red[256 * NEXP];
    const int t   = blockIdx.x;
    const int tid = threadIdx.x;

    float acc[NEXP];
#pragma unroll
    for (int e = 0; e < NEXP; ++e) acc[e] = 0.0f;

    const float4 xv = *(const float4*)(x + (size_t)t * HDIM + tid * 4);
    float xa[4] = { xv.x, xv.y, xv.z, xv.w };
#pragma unroll
    for (int j = 0; j < 4; ++j) {
        const int hh = tid * 4 + j;
        const float4* gp = (const float4*)(gw + (size_t)hh * NEXP);
        float4 g0 = gp[0], g1 = gp[1];
        float xs = xa[j];
        acc[0] += xs * g0.x; acc[1] += xs * g0.y;
        acc[2] += xs * g0.z; acc[3] += xs * g0.w;
        acc[4] += xs * g1.x; acc[5] += xs * g1.y;
        acc[6] += xs * g1.z; acc[7] += xs * g1.w;
    }
#pragma unroll
    for (int e = 0; e < NEXP; ++e) red[tid * NEXP + e] = acc[e];
    __syncthreads();
    for (int s = 128; s > 0; s >>= 1) {
        if (tid < s) {
#pragma unroll
            for (int e = 0; e < NEXP; ++e)
                red[tid * NEXP + e] += red[(tid + s) * NEXP + e];
        }
        __syncthreads();
    }
    if (tid < NEXP) logitsT[(size_t)tid * NTOK + t] = red[tid];
}

// ---------------------------------------------------------------------------
// Kernel 2: per-expert top-k via 4-pass radix select on sortable keys.
// Builds expert_idx[E,k] and inverse map slot_of[E,N] (-1 if unselected).
// Ties at the threshold are taken lowest-index-first (matches jax top_k).
// ---------------------------------------------------------------------------
__global__ __launch_bounds__(1024) void topk_kernel(
    const float* __restrict__ logitsT, int* __restrict__ expert_idx,
    int* __restrict__ slot_of)
{
    __shared__ unsigned skeys[NTOK];
    __shared__ unsigned hist[256];
    __shared__ unsigned sSel[2];
    __shared__ unsigned sCnt[2];

    const int e   = blockIdx.x;
    const int tid = threadIdx.x;

#pragma unroll
    for (int i = 0; i < 8; ++i) {
        const int t = tid + i * 1024;
        float f = logitsT[(size_t)e * NTOK + t];
        unsigned u = __float_as_uint(f);
        skeys[t] = (u & 0x80000000u) ? ~u : (u | 0x80000000u);
        slot_of[e * NTOK + t] = -1;
    }
    __syncthreads();

    unsigned prefix = 0, rem = KSEL;
    for (int b = 3; b >= 0; --b) {
        const int sh = b * 8;
        if (tid < 256) hist[tid] = 0;
        __syncthreads();
#pragma unroll
        for (int i = 0; i < 8; ++i) {
            unsigned key = skeys[tid + i * 1024];
            bool match;
            if (b == 3) match = true;
            else        match = ((key >> (sh + 8)) == (prefix >> (sh + 8)));
            if (match) atomicAdd(&hist[(key >> sh) & 255u], 1u);
        }
        __syncthreads();
        if (tid == 0) {
            unsigned c = 0, selb = 0, r = rem;
            for (int bb = 255; bb >= 0; --bb) {
                unsigned hc = hist[bb];
                if (c + hc >= rem) { selb = (unsigned)bb; r = rem - c; break; }
                c += hc;
            }
            sSel[0] = prefix | (selb << sh);
            sSel[1] = r;
        }
        __syncthreads();
        prefix = sSel[0];
        rem    = sSel[1];
        __syncthreads();
    }
    const unsigned T = prefix;  // k-th largest key

    if (tid == 0) { sCnt[0] = 0; sCnt[1] = 0; }
    __syncthreads();
#pragma unroll
    for (int i = 0; i < 8; ++i) {
        unsigned key = skeys[tid + i * 1024];
        if (key > T) atomicAdd(&sCnt[0], 1u);
    }
    __syncthreads();
    const unsigned cGreater = sCnt[0];
    const unsigned needEq   = KSEL - cGreater;

#pragma unroll
    for (int i = 0; i < 8; ++i) {
        const int t = tid + i * 1024;
        unsigned key = skeys[t];
        if (key > T) {
            unsigned pos = atomicAdd(&sCnt[1], 1u);
            expert_idx[e * KSEL + pos] = t;
            slot_of[e * NTOK + t] = (int)pos;
        } else if (key == T) {
            unsigned rank = 0;                  // rank among equals by index
            for (int j = 0; j < t; ++j) if (skeys[j] == T) ++rank;
            if (rank < needEq) {
                unsigned pos = cGreater + rank;
                expert_idx[e * KSEL + pos] = t;
                slot_of[e * NTOK + t] = (int)pos;
            }
        }
    }
}

// ---------------------------------------------------------------------------
// Kernel 3: software-pipelined, double-buffered bf16 WMMA GEMM (fp32 acc).
// 128x128x32 block tile, 256 threads = 8 waves (2x4); each wave owns a
// 64x32 tile = 4x2 v_wmma_f32_16x16x32_bf16 accumulators.
// Per K-step: issue next tile's global loads -> registers, run ds loads +
// 8 WMMAs on current LDS buffer (hides global latency), then store staged
// registers into the other buffer. One barrier per K-step.
//   GATHER : A rows indirected through expert_idx (GEMM1, A = x fp32)
//   A_BF16 : A already bf16 in memory (GEMM2, A = h)
//   SILU   : apply SiLU and store bf16 (GEMM1), else store fp32 (GEMM2)
// ---------------------------------------------------------------------------
template<bool GATHER, bool A_BF16, bool SILU>
__global__ __launch_bounds__(256) void moe_gemm_kernel(
    const float* __restrict__ Af, const unsigned short* __restrict__ Ab,
    const int* __restrict__ idx, const float* __restrict__ Bw,
    unsigned short* __restrict__ outb, float* __restrict__ outf,
    int K, int lda, int ldb, int ldo,
    long aStrideE, long bStrideE, long oStrideE)
{
    __shared__ unsigned short sA[2][128 * 40];   // [row][K32 + pad8]
    __shared__ unsigned short sB[2][128 * 40];   // transposed: [col][K32 + pad8]
    __shared__ int tok[128];

    const int tid = threadIdx.x;
    const int e   = blockIdx.z;
    const int m0  = blockIdx.y * 128;
    const int n0  = blockIdx.x * 128;

    if (GATHER) { if (tid < 128) tok[tid] = idx[e * KSEL + m0 + tid]; }
    __syncthreads();

    const int lane = tid & 31;
    const int wave = tid >> 5;
    const int wm = wave >> 2;        // 0..1 (M)
    const int wn = wave & 3;         // 0..3 (N)
    const int lr = lane & 15;
    const int lh = lane >> 4;

    // fill-thread mappings
    const int fr = tid >> 1;          // A: row 0..127
    const int fc = (tid & 1) * 16;    // A: col group {0,16}
    const int br = tid >> 3;          // B: K row 0..31
    const int bc = (tid & 7) * 16;    // B: col group 0..112 (contiguous lanes)

    v8f acc[4][2];
#pragma unroll
    for (int mt = 0; mt < 4; ++mt)
#pragma unroll
        for (int nt = 0; nt < 2; ++nt)
#pragma unroll
            for (int i = 0; i < 8; ++i) acc[mt][nt][i] = 0.0f;

    // staged global data
    float4 aF[4];
    uint4  aU[2];
    float4 bF[4];

    auto load_regs = [&](int kk) {
        if (A_BF16) {
            const unsigned short* src =
                Ab + aStrideE * e + (size_t)(m0 + fr) * lda + kk + fc;
            aU[0] = *(const uint4*)(src);
            aU[1] = *(const uint4*)(src + 8);
        } else {
            const int arow = GATHER ? tok[fr] : (m0 + fr);
            const float4* src =
                (const float4*)(Af + aStrideE * e + (size_t)arow * lda + kk + fc);
#pragma unroll
            for (int j = 0; j < 4; ++j) aF[j] = src[j];
        }
        const float4* srcB =
            (const float4*)(Bw + bStrideE * e + (size_t)(kk + br) * ldb + n0 + bc);
#pragma unroll
        for (int j = 0; j < 4; ++j) bF[j] = srcB[j];
    };

    auto store_lds = [&](int buf) {
        if (A_BF16) {
            uint4* dst = (uint4*)(&sA[buf][fr * 40 + fc]);
            dst[0] = aU[0];
            dst[1] = aU[1];
        } else {
            unsigned w[8];
#pragma unroll
            for (int j = 0; j < 4; ++j) {
                w[j * 2 + 0] = f2bfu(aF[j].x) | (f2bfu(aF[j].y) << 16);
                w[j * 2 + 1] = f2bfu(aF[j].z) | (f2bfu(aF[j].w) << 16);
            }
            uint4* dst = (uint4*)(&sA[buf][fr * 40 + fc]);
            dst[0] = make_uint4(w[0], w[1], w[2], w[3]);
            dst[1] = make_uint4(w[4], w[5], w[6], w[7]);
        }
#pragma unroll
        for (int j = 0; j < 4; ++j) {
            sB[buf][(bc + j * 4 + 0) * 40 + br] = (unsigned short)f2bfu(bF[j].x);
            sB[buf][(bc + j * 4 + 1) * 40 + br] = (unsigned short)f2bfu(bF[j].y);
            sB[buf][(bc + j * 4 + 2) * 40 + br] = (unsigned short)f2bfu(bF[j].z);
            sB[buf][(bc + j * 4 + 3) * 40 + br] = (unsigned short)f2bfu(bF[j].w);
        }
    };

    auto compute = [&](int buf) {
        // A 16x32 bf16: lanes 0-15 row M, elems 0-7 = K 0-7, 8-15 = K 16-23;
        //               lanes 16-31 same rows, K 8-15 / 24-31.
        v16bf afr[4];
#pragma unroll
        for (int mt = 0; mt < 4; ++mt) {
            const int row = wm * 64 + mt * 16 + lr;
            v8bf lo = *(const v8bf*)(&sA[buf][row * 40 + lh * 8]);
            v8bf hi = *(const v8bf*)(&sA[buf][row * 40 + lh * 8 + 16]);
            afr[mt] = __builtin_shufflevector(lo, hi,
                0,1,2,3,4,5,6,7,8,9,10,11,12,13,14,15);
        }
        // B 32x16 bf16: lanes 0-15 col N with K 0-7 / 16-23, lanes 16-31 K 8-15 / 24-31.
        v16bf bfr[2];
#pragma unroll
        for (int nt = 0; nt < 2; ++nt) {
            const int col = wn * 32 + nt * 16 + lr;
            v8bf lo = *(const v8bf*)(&sB[buf][col * 40 + lh * 8]);
            v8bf hi = *(const v8bf*)(&sB[buf][col * 40 + lh * 8 + 16]);
            bfr[nt] = __builtin_shufflevector(lo, hi,
                0,1,2,3,4,5,6,7,8,9,10,11,12,13,14,15);
        }
#pragma unroll
        for (int mt = 0; mt < 4; ++mt)
#pragma unroll
            for (int nt = 0; nt < 2; ++nt)
                acc[mt][nt] = __builtin_amdgcn_wmma_f32_16x16x32_bf16(
                    false, afr[mt], false, bfr[nt],
                    (short)0, acc[mt][nt], false, false);
    };

    const int nk = K / 32;
    // prologue: stage and store tile 0
    load_regs(0);
    store_lds(0);
    int buf = 0;
    for (int it = 0; it < nk; ++it) {
        __syncthreads();
        const bool hasNext = (it + 1) < nk;
        if (hasNext) load_regs((it + 1) * 32);   // global loads in flight
        compute(buf);                            // ds loads + 8 WMMAs
        if (hasNext) store_lds(buf ^ 1);         // drain staged regs -> LDS
        buf ^= 1;
    }

    // ---- epilogue: C layout lanes 0-15 -> M rows +0..7, 16-31 -> +8..15 ----
#pragma unroll
    for (int mt = 0; mt < 4; ++mt) {
#pragma unroll
        for (int nt = 0; nt < 2; ++nt) {
            const int grow = m0 + wm * 64 + mt * 16 + lh * 8;
            const int gcol = n0 + wn * 32 + nt * 16 + lr;
#pragma unroll
            for (int v = 0; v < 8; ++v) {
                const float val = acc[mt][nt][v];
                const size_t o = (size_t)(oStrideE * e) + (size_t)(grow + v) * ldo + gcol;
                if (SILU) {
                    // silu(x) = x * sigmoid(x); v_rcp_f32 accuracy >> bf16 ulp
                    const float s =
                        val * __builtin_amdgcn_rcpf(1.0f + __expf(-val));
                    outb[o] = (unsigned short)f2bfu(s);
                } else {
                    outf[o] = val;
                }
            }
        }
    }
}

// ---------------------------------------------------------------------------
// Kernel 4: deterministic combine (fixed expert order, no atomics) + loss=0
// ---------------------------------------------------------------------------
__global__ __launch_bounds__(256) void combine_kernel(
    const float* __restrict__ eo, const int* __restrict__ slot_of,
    float* __restrict__ out)
{
    __shared__ int ss[NEXP];
    const int t   = blockIdx.x;
    const int tid = threadIdx.x;
    if (tid < NEXP) ss[tid] = slot_of[tid * NTOK + t];
    __syncthreads();

    const int col = tid * 4;
    float4 a = make_float4(0.f, 0.f, 0.f, 0.f);
#pragma unroll
    for (int e = 0; e < NEXP; ++e) {
        const int s = ss[e];
        if (s >= 0) {
            float4 v = *(const float4*)(eo + ((size_t)(e * KSEL + s)) * HDIM + col);
            a.x += v.x; a.y += v.y; a.z += v.z; a.w += v.w;
        }
    }
    *(float4*)(out + (size_t)t * HDIM + col) = a;
    if (t == 0 && tid == 0) out[(size_t)NTOK * HDIM] = 0.0f;  // exact lb-loss
}

// ---------------------------------------------------------------------------
extern "C" void kernel_launch(void* const* d_in, const int* in_sizes, int n_in,
                              void* d_out, int out_size, void* d_ws, size_t ws_size,
                              hipStream_t stream)
{
    const float* x  = (const float*)d_in[0];   // [N, H]
    const float* gw = (const float*)d_in[1];   // [H, E]
    const float* w1 = (const float*)d_in[2];   // [E, H, F]
    const float* w2 = (const float*)d_in[3];   // [E, F, H]
    float* out = (float*)d_out;                // [N*H] + 1 loss scalar

    char* ws = (char*)d_ws;
    size_t off = 0;
    auto carve = [&](size_t bytes) {
        size_t o = off;
        off += (bytes + 255) & ~(size_t)255;
        return o;
    };
    float*          logitsT    = (float*)(ws + carve(sizeof(float) * NEXP * NTOK));
    int*            expert_idx = (int*)  (ws + carve(sizeof(int) * NEXP * KSEL));
    int*            slot_of    = (int*)  (ws + carve(sizeof(int) * NEXP * NTOK));
    unsigned short* h          = (unsigned short*)(ws + carve(sizeof(unsigned short) * (size_t)NEXP * KSEL * FDIM));
    float*          eo         = (float*)(ws + carve(sizeof(float) * (size_t)NEXP * KSEL * HDIM));

    gate_logits_kernel<<<NTOK, 256, 0, stream>>>(x, gw, logitsT);
    topk_kernel<<<NEXP, 1024, 0, stream>>>(logitsT, expert_idx, slot_of);

    dim3 g1(FDIM / 128, KSEL / 128, NEXP);   // GEMM1: [k,H] x [H,F] -> silu -> h
    moe_gemm_kernel<true, false, true><<<g1, 256, 0, stream>>>(
        x, nullptr, expert_idx, w1, h, nullptr,
        /*K=*/HDIM, /*lda=*/HDIM, /*ldb=*/FDIM, /*ldo=*/FDIM,
        0L, (long)HDIM * FDIM, (long)KSEL * FDIM);

    dim3 g2(HDIM / 128, KSEL / 128, NEXP);   // GEMM2: [k,F] x [F,H] -> eo
    moe_gemm_kernel<false, true, false><<<g2, 256, 0, stream>>>(
        nullptr, h, nullptr, w2, nullptr, eo,
        /*K=*/FDIM, /*lda=*/FDIM, /*ldb=*/HDIM, /*ldo=*/HDIM,
        (long)KSEL * FDIM, (long)FDIM * HDIM, (long)KSEL * HDIM);

    combine_kernel<<<NTOK, 256, 0, stream>>>(eo, slot_of, out);
}
